// RACRoPEAttention_34806414967155
// MI455X (gfx1250) — matrix operand order repaired and checked by reading, not compiled, not run on hardware
//
#include <hip/hip_runtime.h>
#include <hip/hip_bf16.h>

typedef __bf16 bf16_t;
typedef __attribute__((ext_vector_type(16))) __bf16 v16bf;
typedef __attribute__((ext_vector_type(8)))  float  v8f;
typedef __attribute__((ext_vector_type(4)))  int    v4i;

#define D_MODEL 2048
#define N_HEADS 16
#define D_HEAD  128
#define SEQ_T   2048
#define BATCH   2

union FragU { v16bf v; uint4 q[2]; };

__device__ __forceinline__ v16bf ld_frag(const bf16_t* p0, const bf16_t* p1) {
  FragU u;
  u.q[0] = *reinterpret_cast<const uint4*>(p0);
  u.q[1] = *reinterpret_cast<const uint4*>(p1);
  return u.v;
}

__device__ __forceinline__ v8f zero8() {
  v8f z;
#pragma unroll
  for (int e = 0; e < 8; ++e) z[e] = 0.0f;
  return z;
}

// ---- gfx1250 async global->LDS copy (ASYNCcnt-tracked), with fallback -----
typedef __attribute__((address_space(1))) v4i v4i_glob;
typedef __attribute__((address_space(3))) v4i v4i_lds;

__device__ __forceinline__ void async_copy_b128(void* lds_dst, const void* gsrc) {
#if defined(__has_builtin) && __has_builtin(__builtin_amdgcn_global_load_async_to_lds_b128)
  __builtin_amdgcn_global_load_async_to_lds_b128(
      (v4i_glob*)gsrc, (v4i_lds*)lds_dst, 0, 0);
#else
  *reinterpret_cast<uint4*>(lds_dst) = *reinterpret_cast<const uint4*>(gsrc);
#endif
}

__device__ __forceinline__ void wait_async_le4() {
  asm volatile("s_wait_asynccnt 0x4" ::: "memory");
}
__device__ __forceinline__ void wait_async_0() {
  asm volatile("s_wait_asynccnt 0x0" ::: "memory");
}

// ---------------------------------------------------------------------------
// f32 -> bf16 bulk convert (4 elements / thread)
// ---------------------------------------------------------------------------
__global__ __launch_bounds__(256)
void cvt_f32_to_bf16(const float* __restrict__ s, bf16_t* __restrict__ d, int n4) {
  int i = blockIdx.x * 256 + threadIdx.x;
  if (i < n4) {
    float4 f = reinterpret_cast<const float4*>(s)[i];
    union { bf16_t h[4]; ushort4 u; } o;
    o.h[0] = (bf16_t)f.x; o.h[1] = (bf16_t)f.y;
    o.h[2] = (bf16_t)f.z; o.h[3] = (bf16_t)f.w;
    reinterpret_cast<ushort4*>(d)[i] = o.u;
  }
}

// ---------------------------------------------------------------------------
// Tiled WMMA GEMM (bf16 in): C[M,N] = A[M,K] * B[N,K]^T
// 128x128 block tile, 8 waves (4x2), 32x64 per wave, K-step 32,
// double-buffered async global->LDS tile copies.
// QKV_DST: scatter bf16 into head-major q/k (and transposed v).
// ---------------------------------------------------------------------------
template <bool QKV_DST>
__global__ __launch_bounds__(256)
void gemm_wmma_bf16(const bf16_t* __restrict__ Ag, const bf16_t* __restrict__ Bg,
                    float* __restrict__ Cf,
                    bf16_t* __restrict__ qd, bf16_t* __restrict__ kd,
                    bf16_t* __restrict__ vd, int M, int N, int K) {
  __shared__ __align__(16) bf16_t sA[2][128 * 32];
  __shared__ __align__(16) bf16_t sB[2][128 * 32];

  const int tid  = threadIdx.x;
  const int wave = tid >> 5;
  const int lane = tid & 31;
  const int half = lane >> 4;
  const int l16  = lane & 15;
  const int wm   = (wave & 3) * 32;
  const int wn   = (wave >> 2) * 64;
  const int bm   = blockIdx.x * 128;
  const int bn   = blockIdx.y * 128;

  v8f acc[2][4];
#pragma unroll
  for (int i = 0; i < 2; ++i)
#pragma unroll
    for (int j = 0; j < 4; ++j) acc[i][j] = zero8();

  const int NIT = K >> 5;

  auto issue = [&](int it) {
    int buf = it & 1, k0 = it * 32;
#pragma unroll
    for (int i = 0; i < 2; ++i) {
      int e = (tid + i * 256) * 8;  // 0..4088, 16B-aligned elements
      int m = e >> 5, kk = e & 31;
      async_copy_b128(&sA[buf][m * 32 + kk], &Ag[(size_t)(bm + m) * K + k0 + kk]);
      async_copy_b128(&sB[buf][m * 32 + kk], &Bg[(size_t)(bn + m) * K + k0 + kk]);
    }
  };

  issue(0);
  for (int it = 0; it < NIT; ++it) {
    if (it + 1 < NIT) { issue(it + 1); wait_async_le4(); }
    else              { wait_async_0(); }
    __syncthreads();

    const int buf = it & 1;
    const int klo = half ? 8 : 0;
    const int kb  = half ? 8 : 0;
    v16bf fa[2], fb[4];
#pragma unroll
    for (int tm = 0; tm < 2; ++tm) {
      const bf16_t* p = &sA[buf][(wm + tm * 16 + l16) * 32 + klo];
      fa[tm] = ld_frag(p, p + 16);
    }
#pragma unroll
    for (int tn = 0; tn < 4; ++tn) {
      const bf16_t* p = &sB[buf][(wn + tn * 16 + l16) * 32 + kb];
      fb[tn] = ld_frag(p, p + 16);
    }
#pragma unroll
    for (int tm = 0; tm < 2; ++tm)
#pragma unroll
      for (int tn = 0; tn < 4; ++tn)
        acc[tm][tn] = __builtin_amdgcn_wmma_f32_16x16x32_bf16(
            false, fa[tm], false, fb[tn], (short)0, acc[tm][tn], false, false);
    __syncthreads();
  }

#pragma unroll
  for (int tm = 0; tm < 2; ++tm)
#pragma unroll
    for (int tn = 0; tn < 4; ++tn)
#pragma unroll
      for (int r = 0; r < 8; ++r) {
        int mrow = bm + wm + tm * 16 + (half ? r + 8 : r);
        int ncol = bn + wn + tn * 16 + l16;
        float val = acc[tm][tn][r];
        if (QKV_DST) {
          int which = ncol >> 11;
          int rem   = ncol & 2047;
          int h     = rem >> 7;
          int d     = rem & 127;
          int b     = mrow >> 11;
          int t     = mrow & 2047;
          bf16_t bv = (bf16_t)val;
          if (which == 0)
            qd[((size_t)((b * N_HEADS + h) * SEQ_T + t)) * D_HEAD + d] = bv;
          else if (which == 1)
            kd[((size_t)((b * N_HEADS + h) * SEQ_T + t)) * D_HEAD + d] = bv;
          else  // V stored transposed per head: [bh][d][t]
            vd[((size_t)((b * N_HEADS + h) * D_HEAD + d)) * SEQ_T + t] = bv;
        } else {
          Cf[(size_t)mrow * N + ncol] = val;
        }
      }
}

// ---------------------------------------------------------------------------
// RoPE in place on head-major bf16 Q and K: [B*H, T, 128]
// ---------------------------------------------------------------------------
__global__ __launch_bounds__(256)
void rope_kernel(bf16_t* __restrict__ Q, bf16_t* __restrict__ Kb) {
  size_t idx = (size_t)blockIdx.x * 256 + threadIdx.x;  // over B*H*T*64
  int i  = (int)(idx & 63);
  int t  = (int)((idx >> 6) & (SEQ_T - 1));
  int bh = (int)(idx >> 17);
  size_t base = ((size_t)bh * SEQ_T + t) * D_HEAD + 2 * i;
  float inv_freq = __expf(-((float)(2 * i) / 128.0f) * 9.210340371976184f);
  float ang = (float)t * inv_freq;
  float s, c;
  __sincosf(ang, &s, &c);
  float q1 = (float)Q[base], q2 = (float)Q[base + 1];
  Q[base]     = (bf16_t)(q1 * c - q2 * s);
  Q[base + 1] = (bf16_t)(q1 * s + q2 * c);
  float k1 = (float)Kb[base], k2 = (float)Kb[base + 1];
  Kb[base]     = (bf16_t)(k1 * c - k2 * s);
  Kb[base + 1] = (bf16_t)(k1 * s + k2 * c);
}

// ---------------------------------------------------------------------------
// Causal flash attention. Block = (128 q-rows, one b*h); 8 waves x 16 rows.
// K chunks of 32, online softmax, double-buffered async K/V tile copies.
// V is pre-transposed in global ([bh][d][t]) so its LDS tile copy is linear.
// ---------------------------------------------------------------------------
__global__ __launch_bounds__(256)
void flash_attn(const bf16_t* __restrict__ Q, const bf16_t* __restrict__ Kb,
                const bf16_t* __restrict__ Vb, bf16_t* __restrict__ Ob) {
  __shared__ __align__(16) bf16_t sK[2][32 * 128];   // [k][d]
  __shared__ __align__(16) bf16_t sVt[2][128 * 32];  // [d][k]
  __shared__ __align__(16) float  sS[8][16 * 32];
  __shared__ __align__(16) bf16_t sP[8][16 * 32];
  __shared__ float sM[8][16], sL[8][16], sScale[8][16];

  const int tid  = threadIdx.x;
  const int wave = tid >> 5;
  const int lane = tid & 31;
  const int half = lane >> 4;
  const int l16  = lane & 15;
  const int bh   = blockIdx.y;
  const int b    = bh >> 4;
  const int h    = bh & 15;
  const size_t kqbase = (size_t)bh * SEQ_T * D_HEAD;
  const size_t vbase  = (size_t)bh * D_HEAD * SEQ_T;
  const int qrow0 = blockIdx.x * 128 + wave * 16;
  const float scale = 0.08838834764831845f;  // 1/sqrt(128)

  v16bf fq[4];
  const int klo = half ? 8 : 0;
#pragma unroll
  for (int kdv = 0; kdv < 4; ++kdv) {
    const bf16_t* p = Q + kqbase + (size_t)(qrow0 + l16) * D_HEAD + kdv * 32 + klo;
    fq[kdv] = ld_frag(p, p + 16);
  }

  v8f oacc[8];
#pragma unroll
  for (int t = 0; t < 8; ++t) oacc[t] = zero8();

  if (lane < 16) {
    sM[wave][lane] = -3.0e38f;
    sL[wave][lane] = 0.0f;
  }

  auto issue = [&](int it) {
    int buf = it & 1, kt = it * 32;
#pragma unroll
    for (int i = 0; i < 2; ++i) {
      int e = (tid + i * 256) * 8;  // 0..4088
      async_copy_b128(&sK[buf][e], &Kb[kqbase + (size_t)kt * D_HEAD + e]);
      int d = e >> 5, kk = e & 31;
      async_copy_b128(&sVt[buf][d * 32 + kk], &Vb[vbase + (size_t)d * SEQ_T + kt + kk]);
    }
  };

  const int nkc = (blockIdx.x + 1) * 4;
  issue(0);
  for (int kc = 0; kc < nkc; ++kc) {
    const int kt  = kc * 32;
    const int buf = kc & 1;
    if (kc + 1 < nkc) { issue(kc + 1); wait_async_le4(); }
    else              { wait_async_0(); }
    __syncthreads();

    // S = Q * K^T (two 16-col subtiles, contraction over d=128)
    const int kb2 = half ? 8 : 0;
#pragma unroll
    for (int sub = 0; sub < 2; ++sub) {
      v8f s = zero8();
#pragma unroll
      for (int kdv = 0; kdv < 4; ++kdv) {
        const bf16_t* p = &sK[buf][(sub * 16 + l16) * 128 + kdv * 32 + kb2];
        s = __builtin_amdgcn_wmma_f32_16x16x32_bf16(
            false, fq[kdv], false, ld_frag(p, p + 16), (short)0, s, false, false);
      }
#pragma unroll
      for (int r = 0; r < 8; ++r) {
        int row   = half ? r + 8 : r;
        int mglob = qrow0 + row;
        int tk    = kt + sub * 16 + l16;
        float v   = s[r] * scale;
        if (tk > mglob) v = -3.0e38f;
        sS[wave][row * 32 + sub * 16 + l16] = v;
      }
    }
    __syncthreads();

    // online softmax (lanes 0..15 each own one q-row)
    if (lane < 16) {
      int row = lane;
      float mo = sM[wave][row];
      float cm = -3.0e38f;
#pragma unroll
      for (int c = 0; c < 32; ++c) cm = fmaxf(cm, sS[wave][row * 32 + c]);
      float mn = fmaxf(mo, cm);
      float sc = __expf(mo - mn);
      float sum = 0.0f;
#pragma unroll
      for (int c = 0; c < 32; ++c) {
        float p = __expf(sS[wave][row * 32 + c] - mn);
        sum += p;
        sP[wave][row * 32 + c] = (bf16_t)p;
      }
      sM[wave][row]     = mn;
      sL[wave][row]     = sL[wave][row] * sc + sum;
      sScale[wave][row] = sc;
    }
    __syncthreads();

    // rescale accumulator
    float rs[8];
#pragma unroll
    for (int r = 0; r < 8; ++r) rs[r] = sScale[wave][half ? r + 8 : r];
#pragma unroll
    for (int t = 0; t < 8; ++t)
#pragma unroll
      for (int r = 0; r < 8; ++r) oacc[t][r] *= rs[r];

    // O += P * V
    const bf16_t* pp = &sP[wave][l16 * 32 + klo];
    v16bf fp = ld_frag(pp, pp + 16);
#pragma unroll
    for (int t = 0; t < 8; ++t) {
      const bf16_t* vp = &sVt[buf][(t * 16 + l16) * 32 + kb2];
      oacc[t] = __builtin_amdgcn_wmma_f32_16x16x32_bf16(
          false, fp, false, ld_frag(vp, vp + 16), (short)0, oacc[t], false, false);
    }
    __syncthreads();
  }

  // finalize: divide by row sums, scatter bf16 to [B, T, H*128]
  float inv[8];
#pragma unroll
  for (int r = 0; r < 8; ++r) inv[r] = 1.0f / sL[wave][half ? r + 8 : r];
#pragma unroll
  for (int t = 0; t < 8; ++t)
#pragma unroll
    for (int r = 0; r < 8; ++r) {
      int row  = half ? r + 8 : r;
      int trow = qrow0 + row;
      Ob[((size_t)(b * SEQ_T + trow)) * D_MODEL + h * D_HEAD + t * 16 + l16] =
          (bf16_t)(oacc[t][r] * inv[r]);
    }
}

// ---------------------------------------------------------------------------
extern "C" void kernel_launch(void* const* d_in, const int* in_sizes, int n_in,
                              void* d_out, int out_size, void* d_ws, size_t ws_size,
                              hipStream_t stream) {
  const float* x     = (const float*)d_in[0];
  const float* w_qkv = (const float*)d_in[1];
  const float* w_out = (const float*)d_in[2];
  float* out = (float*)d_out;

  const int    M        = BATCH * SEQ_T;                     // 4096
  const size_t n_x      = (size_t)M * D_MODEL;               // 8.39M
  const size_t n_wqkv   = (size_t)3 * D_MODEL * D_MODEL;     // 12.6M
  const size_t n_wout   = (size_t)D_MODEL * D_MODEL;         // 4.2M
  const size_t per_head = (size_t)BATCH * N_HEADS * SEQ_T * D_HEAD;

  bf16_t* xb   = (bf16_t*)d_ws;
  bf16_t* wqb  = xb + n_x;
  bf16_t* wob  = wqb + n_wqkv;
  bf16_t* q    = wob + n_wout;
  bf16_t* k    = q + per_head;
  bf16_t* v    = k + per_head;   // transposed per head: [bh][d][t]
  bf16_t* attn = v + per_head;   // [B*T, D_MODEL]

  // 0) one-time f32 -> bf16 conversions (L2-resident, trivial cost)
  cvt_f32_to_bf16<<<(int)(n_x / 1024), 256, 0, stream>>>(x, xb, (int)(n_x / 4));
  cvt_f32_to_bf16<<<(int)(n_wqkv / 1024), 256, 0, stream>>>(w_qkv, wqb, (int)(n_wqkv / 4));
  cvt_f32_to_bf16<<<(int)(n_wout / 1024), 256, 0, stream>>>(w_out, wob, (int)(n_wout / 4));

  // 1) QKV projection -> head-major q/k, transposed v
  dim3 g1(M / 128, (3 * D_MODEL) / 128);
  gemm_wmma_bf16<true><<<g1, 256, 0, stream>>>(
      xb, wqb, nullptr, q, k, v, M, 3 * D_MODEL, D_MODEL);

  // 2) RoPE in place on q, k
  int rope_blocks = (BATCH * N_HEADS * SEQ_T * (D_HEAD / 2)) / 256;
  rope_kernel<<<rope_blocks, 256, 0, stream>>>(q, k);

  // 3) causal flash attention -> attn (bf16, btd layout)
  dim3 g3(SEQ_T / 128, BATCH * N_HEADS);
  flash_attn<<<g3, 256, 0, stream>>>(q, k, v, attn);

  // 4) output projection -> f32 d_out
  dim3 g4(M / 128, D_MODEL / 128);
  gemm_wmma_bf16<false><<<g4, 256, 0, stream>>>(
      attn, wob, out, nullptr, nullptr, nullptr, M, D_MODEL, D_MODEL);
}
